// Block_57921928954115
// MI455X (gfx1250) — matrix operand order repaired and checked by reading, not compiled, not run on hardware
//
#include <hip/hip_runtime.h>

// ---------------------------------------------------------------------------
// Transformer block (RMSNorm -> MHA+RoPE -> +res -> RMSNorm -> SwiGLU FF -> +res)
// for MI455X (gfx1250): bf16 WMMA + TDM (tensor_load_to_lds) staged GEMMs.
// ---------------------------------------------------------------------------

constexpr int B_ = 4, T_ = 1024, C_ = 768, H_ = 12, S_ = 64;
constexpr int FF_ = 3072, HID_ = 2048;
constexpr int M_ = B_ * T_;           // 4096 rows (b,t)
constexpr float EPS_ = 1e-6f;

typedef __attribute__((ext_vector_type(16))) __bf16 v16bf;
typedef __attribute__((ext_vector_type(8)))  float  v8f;
typedef __attribute__((ext_vector_type(4)))  unsigned int v4u;
typedef __attribute__((ext_vector_type(8)))  int v8i;
typedef __attribute__((ext_vector_type(4)))  int v4i;

union Frag16 { v16bf v; __bf16 e[16]; uint4 u4[2]; };

#if defined(__has_builtin)
#if __has_builtin(__builtin_amdgcn_tensor_load_to_lds)
#define HAVE_TDM 1
#endif
#endif
#ifndef HAVE_TDM
#define HAVE_TDM 0
#endif

// ---------------------------------------------------------------------------
// TDM: DMA a (tile_h x tile_w) bf16 tile (row stride = stride_el) from global
// into LDS at lds_ptr. Descriptor per CDNA5 ISA.
// Issued once by a single wave; completion tracked with TENSORcnt.
// ---------------------------------------------------------------------------
__device__ inline void tdm_load_tile(const __bf16* g, __bf16* lds_ptr,
                                     int tile_w_el, int tile_h, int stride_el) {
#if HAVE_TDM
    unsigned long long ga = (unsigned long long)(uintptr_t)g;
    unsigned int lds_off = (unsigned int)(uintptr_t)lds_ptr;  // flat[31:0] = LDS offset
    v4u g0 = {0u, 0u, 0u, 0u};
    g0.x = 1u;                                           // count=1, user mode
    g0.y = lds_off;                                      // lds_addr (bytes)
    g0.z = (unsigned int)ga;                             // global_addr[31:0]
    g0.w = ((unsigned int)(ga >> 32) & 0x01FFFFFFu)      // global_addr[56:32]
         | (2u << 30);                                   // type=2 ("image")
    unsigned int td0 = (unsigned int)tile_w_el;          // tensor_dim0
    unsigned int td1 = (unsigned int)tile_h;             // tensor_dim1
    unsigned long long st0 = (unsigned long long)stride_el;
    v8i g1 = {0, 0, 0, 0, 0, 0, 0, 0};
    g1[0] = (int)(1u << 16);                             // data_size=1 -> 2 bytes
    g1[1] = (int)((td0 & 0xFFFFu) << 16);                // tensor_dim0[15:0]
    g1[2] = (int)((td0 >> 16) | ((td1 & 0xFFFFu) << 16));// td0[31:16] | td1[15:0]
    g1[3] = (int)((td1 >> 16) | ((unsigned int)tile_w_el << 16)); // tile_dim0
    g1[4] = (int)((unsigned int)tile_h & 0xFFFFu);       // tile_dim1
    g1[5] = (int)(unsigned int)(st0 & 0xFFFFFFFFu);      // tensor_dim0_stride lo
    g1[6] = (int)(unsigned int)((st0 >> 32) & 0xFFFFu);  // tensor_dim0_stride hi
    v4i gz = {0, 0, 0, 0};
#if __clang_major__ >= 23
    v8i gz8 = {0, 0, 0, 0, 0, 0, 0, 0};
    __builtin_amdgcn_tensor_load_to_lds(g0, g1, gz, gz, gz8, 0);
#else
    __builtin_amdgcn_tensor_load_to_lds(g0, g1, gz, gz, 0);
#endif
#else
    // Fallback (no TDM builtin): the calling wave copies the tile row-by-row.
    const int lane = threadIdx.x & 31;
    for (int r = lane; r < tile_h; r += 32) {
        const uint4* src = (const uint4*)(g + (size_t)r * stride_el);
        uint4* dd = (uint4*)(lds_ptr + (size_t)r * tile_w_el);
        for (int c = 0; c < tile_w_el / 8; ++c) dd[c] = src[c];
    }
#endif
}

// ---------------------------------------------------------------------------
// Fragment loaders (contiguous k order: lane half h holds k = 16h..16h+15, wave32)
// ---------------------------------------------------------------------------
__device__ inline v16bf load_afrag(const __bf16* base /*at (m0,k0)*/, int lda, int lane) {
    const int row = lane & 15, half = lane >> 4;
    const __bf16* p = base + (size_t)row * lda + half * 16;
    Frag16 f;
    f.u4[0] = *(const uint4*)(p);
    f.u4[1] = *(const uint4*)(p + 8);
    return f.v;
}

__device__ inline v16bf load_bfrag(const __bf16* bt /*at (n0,k0), [N][K]*/, int ldb, int lane) {
    const int col = lane & 15, half = lane >> 4;
    const __bf16* p = bt + (size_t)col * ldb + half * 16;
    Frag16 f;
    f.u4[0] = *(const uint4*)(p);
    f.u4[1] = *(const uint4*)(p + 8);
    return f.v;
}

__device__ inline v8f wmma_bf16(v16bf a, v16bf b, v8f c) {
    return __builtin_amdgcn_wmma_f32_16x16x32_bf16(false, a, false, b, (short)0, c, false, false);
}

// ---------------------------------------------------------------------------
// Small utility kernels
// ---------------------------------------------------------------------------
__global__ void k_zero(float* p, int n) {
    int i = blockIdx.x * blockDim.x + threadIdx.x;
    if (i < n) p[i] = 0.f;
}

__global__ void k_wt2d(const float* in, __bf16* out, int K, int N) {
    size_t i = (size_t)blockIdx.x * blockDim.x + threadIdx.x;
    if (i >= (size_t)K * N) return;
    int k = (int)(i / N), n = (int)(i % N);
    out[(size_t)n * K + k] = (__bf16)in[i];
}

__global__ void k_wtqkv(const float* in, __bf16* out) {
    size_t i = (size_t)blockIdx.x * blockDim.x + threadIdx.x;
    if (i >= (size_t)H_ * C_ * S_) return;
    int s  = (int)(i % S_);
    int c  = (int)((i / S_) % C_);
    int hh = (int)(i / ((size_t)C_ * S_));
    out[((size_t)hh * S_ + s) * C_ + c] = (__bf16)in[i];
}

__global__ void k_sumsq(const float* x, long n, float* slot) {
    __shared__ float sh[8];
    float s = 0.f;
    for (long i = (long)blockIdx.x * blockDim.x + threadIdx.x; i < n;
         i += (long)gridDim.x * blockDim.x) {
        float v = x[i];
        s += v * v;
    }
    for (int d = 16; d; d >>= 1) s += __shfl_xor(s, d, 32);
    if ((threadIdx.x & 31) == 0) sh[threadIdx.x >> 5] = s;
    __syncthreads();
    if (threadIdx.x == 0) {
        float t = 0.f;
        for (int w = 0; w < (int)(blockDim.x >> 5); ++w) t += sh[w];
        atomicAdd(slot, t);
    }
}

__global__ void k_rmsapply(const float* x, const float* gamma, __bf16* h,
                           const float* slot, long n, long ntot) {
    long i = (long)blockIdx.x * blockDim.x + threadIdx.x;
    if (i >= n) return;
    float inv = rsqrtf(EPS_ + slot[0] / (float)ntot);
    h[i] = (__bf16)(x[i] * inv * gamma[i % C_]);
}

__global__ void k_rope(__bf16* p) {
    long i = (long)blockIdx.x * blockDim.x + threadIdx.x;
    if (i >= (long)M_ * H_ * (S_ / 2)) return;
    int s2 = (int)(i & 31);
    long row = i >> 5;                    // (b,t,h) flat
    int t = (int)((row / H_) % T_);
    float theta = __powf(10000.f, -(float)s2 / 16.f);
    float ang = (float)t * theta;
    float cs, sn;
    __sincosf(ang, &sn, &cs);
    __bf16* q = p + row * S_ + 2 * s2;
    float ev = (float)q[0], od = (float)q[1];
    q[0] = (__bf16)(ev * cs - od * sn);
    q[1] = (__bf16)(od * cs + ev * sn);
}

// ---------------------------------------------------------------------------
// WMMA GEMM with TDM-staged, double-buffered A panel in LDS.
// Grid: (N/256, M/32). Block: 256 threads = 8 waves as 2(M) x 4(N);
// each wave computes 16x64 with A-frags from LDS, B-frags from global [N][K].
// EPI: 0=store bf16  1=+bias bf16  2=res f32  3=+bias res f32
//      4=swiglu: out = bf16( aux * g * sigmoid(g) ), g = acc + bias
// ---------------------------------------------------------------------------
template <int EPI>
__global__ void k_gemm(const __bf16* __restrict__ A, const __bf16* __restrict__ Bt,
                       const float* __restrict__ bias, const float* __restrict__ res,
                       const __bf16* __restrict__ aux, void* __restrict__ out,
                       int M, int N, int K) {
    __shared__ __align__(16) __bf16 sA[2][32 * 64];   // double-buffered A panel

    const int lane = threadIdx.x & 31;
    const int w = threadIdx.x >> 5;
    const int mtl = w >> 2;          // 0..1
    const int ntl = w & 3;           // 0..3
    const int m0 = blockIdx.y * 32;
    const int n0 = blockIdx.x * 256 + ntl * 64;
    const int mrow0 = m0 + mtl * 16;

    v8f acc[4] = {};
    const int nk = K >> 6;           // K/64 panels

    if (w == 0) {
        tdm_load_tile(A + (size_t)m0 * K, &sA[0][0], 64, 32, K);
        __builtin_amdgcn_s_wait_tensorcnt(0);
    }
    __syncthreads();

    for (int it = 0; it < nk; ++it) {
        const __bf16* aTile = &sA[it & 1][0] + mtl * 16 * 64;
        if (w == 0 && it + 1 < nk)
            tdm_load_tile(A + (size_t)m0 * K + (size_t)(it + 1) * 64,
                          &sA[(it + 1) & 1][0], 64, 32, K);
#pragma unroll
        for (int kk = 0; kk < 64; kk += 32) {
            v16bf a = load_afrag(aTile + kk, 64, lane);
#pragma unroll
            for (int j = 0; j < 4; ++j) {
                const __bf16* bp = Bt + (size_t)(n0 + j * 16) * K + it * 64 + kk;
                __builtin_prefetch(bp + 64, 0, 0);
                v16bf b = load_bfrag(bp, K, lane);
                acc[j] = wmma_bf16(a, b, acc[j]);
            }
        }
        if (w == 0 && it + 1 < nk) __builtin_amdgcn_s_wait_tensorcnt(0);
        __syncthreads();
    }

    const int col = lane & 15, rbase = (lane >> 4) * 8;
#pragma unroll
    for (int j = 0; j < 4; ++j) {
        const int n = n0 + j * 16 + col;
        const float bv = (EPI == 1 || EPI == 3 || EPI == 4) ? bias[n] : 0.f;
#pragma unroll
        for (int i = 0; i < 8; ++i) {
            const size_t idx = (size_t)(mrow0 + rbase + i) * N + n;
            float g = acc[j][i] + bv;
            if (EPI == 2 || EPI == 3) {
                ((float*)out)[idx] = res[idx] + g;
            } else if (EPI == 4) {
                float wv = (float)aux[idx];
                ((__bf16*)out)[idx] = (__bf16)(wv * g / (1.f + __expf(-g)));
            } else {
                ((__bf16*)out)[idx] = (__bf16)g;
            }
        }
    }
}

// ---------------------------------------------------------------------------
// Flash attention. q,k,v,o are bf16 [b,t,h,s]. One wave: 16 queries x S=64.
// Key blocks of 32; online softmax; QK^T and P.V via WMMA; P and V staged in LDS.
// Scores scaled by C^-0.5 (faithful to reference).
// ---------------------------------------------------------------------------
constexpr int AW_ = 4;  // waves per block
__global__ void k_attn(const __bf16* __restrict__ q, const __bf16* __restrict__ k,
                       const __bf16* __restrict__ v, __bf16* __restrict__ o) {
    __shared__ __align__(16) __bf16 pst[AW_][16 * 32];
    __shared__ __align__(16) __bf16 vst[AW_][32 * 64];

    const int lane = threadIdx.x & 31;
    const int w = threadIdx.x >> 5;
    const int gw = blockIdx.x * AW_ + w;
    const int qt = gw % (T_ / 16);
    const int hh = (gw / (T_ / 16)) % H_;
    const int b = gw / ((T_ / 16) * H_);
    const int qbase = qt * 16;
    const int lda = H_ * S_;
    const float scale = rsqrtf((float)C_);
    const float NEG = -1e30f;

    const __bf16* qbaseptr = q + ((size_t)(b * T_ + qbase) * H_ + hh) * S_;
    v16bf aq0 = load_afrag(qbaseptr, lda, lane);
    v16bf aq1 = load_afrag(qbaseptr + 32, lda, lane);

    v8f O[4] = {};
    float mrow[8], lrow[8];
#pragma unroll
    for (int i = 0; i < 8; ++i) { mrow[i] = NEG; lrow[i] = 0.f; }

    const int col = lane & 15;
    const int rhalf = lane >> 4;  // 0: rows 0-7, 1: rows 8-15

    for (int kb = 0; kb < qbase + 16; kb += 32) {
        // ---- stage V block (32 keys x 64) into LDS, one row per lane ----
        {
            const int key = kb + lane;
            uint4* dst = (uint4*)&vst[w][lane * 64];
            if (key < T_) {
                const uint4* src = (const uint4*)(v + ((size_t)(b * T_ + key) * H_ + hh) * S_);
#pragma unroll
                for (int c4 = 0; c4 < 8; ++c4) dst[c4] = src[c4];
            } else {
                uint4 z = {0u, 0u, 0u, 0u};
#pragma unroll
                for (int c4 = 0; c4 < 8; ++c4) dst[c4] = z;
            }
        }

        // ---- scores S (16 queries x 32 keys) via WMMA ----
        v8f Sc[2] = {};
#pragma unroll
        for (int ns = 0; ns < 2; ++ns) {
            const int key = kb + ns * 16 + col;
            Frag16 f0, f1;
            if (key < T_) {
                const __bf16* kp = k + ((size_t)(b * T_ + key) * H_ + hh) * S_ + rhalf * 16;
                f0.u4[0] = *(const uint4*)(kp);
                f0.u4[1] = *(const uint4*)(kp + 8);
                f1.u4[0] = *(const uint4*)(kp + 32);
                f1.u4[1] = *(const uint4*)(kp + 40);
            } else {
                uint4 z = {0u, 0u, 0u, 0u};
                f0.u4[0] = f0.u4[1] = f1.u4[0] = f1.u4[1] = z;
            }
            Sc[ns] = wmma_bf16(aq0, f0.v, Sc[ns]);
            Sc[ns] = wmma_bf16(aq1, f1.v, Sc[ns]);
        }

        // ---- scale + causal mask + online softmax ----
        float P0[8], P1[8];
#pragma unroll
        for (int i = 0; i < 8; ++i) {
            const int qrow = qbase + rhalf * 8 + i;
            float s0 = Sc[0][i] * scale, s1 = Sc[1][i] * scale;
            if (kb + col > qrow) s0 = NEG;
            if (kb + 16 + col > qrow) s1 = NEG;
            float tm = fmaxf(s0, s1);
            for (int d = 1; d < 16; d <<= 1) tm = fmaxf(tm, __shfl_xor(tm, d, 32));
            float mnew = fmaxf(mrow[i], tm);
            float corr = __expf(mrow[i] - mnew);
            float p0 = __expf(s0 - mnew), p1 = __expf(s1 - mnew);
            float ts = p0 + p1;
            for (int d = 1; d < 16; d <<= 1) ts += __shfl_xor(ts, d, 32);
            lrow[i] = lrow[i] * corr + ts;
            mrow[i] = mnew;
#pragma unroll
            for (int j = 0; j < 4; ++j) O[j][i] *= corr;
            P0[i] = p0;
            P1[i] = p1;
        }

        // ---- write P tile (16x32) to LDS, reload as A-frag ----
#pragma unroll
        for (int i = 0; i < 8; ++i) {
            const int r = rhalf * 8 + i;
            pst[w][r * 32 + col]      = (__bf16)P0[i];
            pst[w][r * 32 + 16 + col] = (__bf16)P1[i];
        }
        asm volatile("s_wait_dscnt 0" ::: "memory");

        v16bf aP = load_afrag(&pst[w][0], 32, lane);

        // ---- O += P x V via WMMA (B-frag gathered from staged V) ----
#pragma unroll
        for (int j = 0; j < 4; ++j) {
            Frag16 fb;
#pragma unroll
            for (int e = 0; e < 16; ++e)
                fb.e[e] = vst[w][(rhalf * 16 + e) * 64 + j * 16 + col];
            O[j] = wmma_bf16(aP, fb.v, O[j]);
        }
        asm volatile("s_wait_dscnt 0" ::: "memory");
    }

    // ---- normalize and store [b,t,h,s] ----
#pragma unroll
    for (int j = 0; j < 4; ++j) {
#pragma unroll
        for (int i = 0; i < 8; ++i) {
            const int r = rhalf * 8 + i;
            o[((size_t)(b * T_ + qbase + r) * H_ + hh) * S_ + j * 16 + col] =
                (__bf16)(O[j][i] / lrow[i]);
        }
    }
}

// ---------------------------------------------------------------------------
// Host side
// ---------------------------------------------------------------------------
static inline size_t alignup(size_t x) { return (x + 255) & ~(size_t)255; }

extern "C" void kernel_launch(void* const* d_in, const int* in_sizes, int n_in,
                              void* d_out, int out_size, void* d_ws, size_t ws_size,
                              hipStream_t stream) {
    const float* x  = (const float*)d_in[0];
    const float* Wq = (const float*)d_in[1];
    const float* Wk = (const float*)d_in[2];
    const float* Wv = (const float*)d_in[3];
    const float* Wo = (const float*)d_in[4];
    const float* g1 = (const float*)d_in[5];
    const float* g2 = (const float*)d_in[6];
    const float* W1 = (const float*)d_in[7];
    const float* b1 = (const float*)d_in[8];
    const float* Ww = (const float*)d_in[9];
    const float* bw = (const float*)d_in[10];
    const float* Wg = (const float*)d_in[11];
    const float* bg = (const float*)d_in[12];
    const float* Wd = (const float*)d_in[13];
    const float* bd = (const float*)d_in[14];
    const float* W2 = (const float*)d_in[15];
    const float* b2 = (const float*)d_in[16];
    float* out = (float*)d_out;

    char* ws = (char*)d_ws;
    size_t cur = 0;
    auto alloc = [&](size_t bytes) { void* p = ws + cur; cur = alignup(cur + bytes); return p; };

    float*  red  = (float*)alloc(8 * sizeof(float));
    __bf16* wqt  = (__bf16*)alloc((size_t)H_ * S_ * C_ * 2);
    __bf16* wkt  = (__bf16*)alloc((size_t)H_ * S_ * C_ * 2);
    __bf16* wvt  = (__bf16*)alloc((size_t)H_ * S_ * C_ * 2);
    __bf16* wot  = (__bf16*)alloc((size_t)C_ * C_ * 2);
    __bf16* w1t  = (__bf16*)alloc((size_t)FF_ * C_ * 2);
    __bf16* wwt  = (__bf16*)alloc((size_t)HID_ * FF_ * 2);
    __bf16* wgt  = (__bf16*)alloc((size_t)HID_ * FF_ * 2);
    __bf16* wdt  = (__bf16*)alloc((size_t)FF_ * HID_ * 2);
    __bf16* w2t  = (__bf16*)alloc((size_t)C_ * FF_ * 2);
    __bf16* h1   = (__bf16*)alloc((size_t)M_ * C_ * 2);   // reused as h2
    __bf16* qb   = (__bf16*)alloc((size_t)M_ * C_ * 2);
    __bf16* kbuf = (__bf16*)alloc((size_t)M_ * C_ * 2);
    __bf16* vb   = (__bf16*)alloc((size_t)M_ * C_ * 2);
    __bf16* ao   = (__bf16*)alloc((size_t)M_ * C_ * 2);
    float*  out1 = (float*)alloc((size_t)M_ * C_ * 4);
    __bf16* ub   = (__bf16*)alloc((size_t)M_ * FF_ * 2);  // reused as ff1
    __bf16* wx   = (__bf16*)alloc((size_t)M_ * HID_ * 2);
    __bf16* sact = (__bf16*)alloc((size_t)M_ * HID_ * 2);
    (void)ws_size; (void)n_in; (void)in_sizes; (void)out_size;

    auto cdiv = [](long a, long b) { return (int)((a + b - 1) / b); };

    k_zero<<<1, 32, 0, stream>>>(red, 8);

    {
        int n3 = H_ * C_ * S_;
        k_wtqkv<<<cdiv(n3, 256), 256, 0, stream>>>(Wq, wqt);
        k_wtqkv<<<cdiv(n3, 256), 256, 0, stream>>>(Wk, wkt);
        k_wtqkv<<<cdiv(n3, 256), 256, 0, stream>>>(Wv, wvt);
        k_wt2d<<<cdiv((long)C_ * C_, 256), 256, 0, stream>>>(Wo, wot, C_, C_);
        k_wt2d<<<cdiv((long)C_ * FF_, 256), 256, 0, stream>>>(W1, w1t, C_, FF_);
        k_wt2d<<<cdiv((long)FF_ * HID_, 256), 256, 0, stream>>>(Ww, wwt, FF_, HID_);
        k_wt2d<<<cdiv((long)FF_ * HID_, 256), 256, 0, stream>>>(Wg, wgt, FF_, HID_);
        k_wt2d<<<cdiv((long)HID_ * FF_, 256), 256, 0, stream>>>(Wd, wdt, HID_, FF_);
        k_wt2d<<<cdiv((long)FF_ * C_, 256), 256, 0, stream>>>(W2, w2t, FF_, C_);
    }

    const long nx = (long)M_ * C_;

    // rms1 -> h1
    k_sumsq<<<1024, 256, 0, stream>>>(x, nx, red + 0);
    k_rmsapply<<<cdiv(nx, 256), 256, 0, stream>>>(x, g1, h1, red + 0, nx, nx);

    auto ggrid = [](int M, int N) { return dim3((unsigned)(N / 256), (unsigned)(M / 32), 1); };

    // QKV projections (M=4096, N=768, K=768)
    k_gemm<0><<<ggrid(M_, C_), 256, 0, stream>>>(h1, wqt, nullptr, nullptr, nullptr, qb, M_, C_, C_);
    k_gemm<0><<<ggrid(M_, C_), 256, 0, stream>>>(h1, wkt, nullptr, nullptr, nullptr, kbuf, M_, C_, C_);
    k_gemm<0><<<ggrid(M_, C_), 256, 0, stream>>>(h1, wvt, nullptr, nullptr, nullptr, vb, M_, C_, C_);

    // RoPE on q, k
    {
        long nr = (long)M_ * H_ * (S_ / 2);
        k_rope<<<cdiv(nr, 256), 256, 0, stream>>>(qb);
        k_rope<<<cdiv(nr, 256), 256, 0, stream>>>(kbuf);
    }

    // attention
    {
        int waves = B_ * H_ * (T_ / 16);
        k_attn<<<waves / AW_, AW_ * 32, 0, stream>>>(qb, kbuf, vb, ao);
    }

    // out1 = x + attn @ Wo
    k_gemm<2><<<ggrid(M_, C_), 256, 0, stream>>>(ao, wot, nullptr, x, nullptr, out1, M_, C_, C_);

    // rms2 -> h2 (reuse h1)
    k_sumsq<<<1024, 256, 0, stream>>>(out1, nx, red + 1);
    k_rmsapply<<<cdiv(nx, 256), 256, 0, stream>>>(out1, g2, h1, red + 1, nx, nx);

    // u = h2 @ W1 + b1
    k_gemm<1><<<ggrid(M_, FF_), 256, 0, stream>>>(h1, w1t, b1, nullptr, nullptr, ub, M_, FF_, C_);
    // wx = u @ Ww + bw
    k_gemm<1><<<ggrid(M_, HID_), 256, 0, stream>>>(ub, wwt, bw, nullptr, nullptr, wx, M_, HID_, FF_);
    // sact = wx * silu(u @ Wg + bg)
    k_gemm<4><<<ggrid(M_, HID_), 256, 0, stream>>>(ub, wgt, bg, nullptr, wx, sact, M_, HID_, FF_);
    // ff1 = sact @ Wd + bd (reuse ub)
    k_gemm<1><<<ggrid(M_, FF_), 256, 0, stream>>>(sact, wdt, bd, nullptr, nullptr, ub, M_, FF_, HID_);
    // out = out1 + ff1 @ W2 + b2
    k_gemm<3><<<ggrid(M_, C_), 256, 0, stream>>>(ub, w2t, b2, out1, nullptr, out, M_, C_, FF_);
}
